// RNN_59210419142980
// MI455X (gfx1250) — matrix operand order repaired and checked by reading, not compiled, not run on hardware
//
#include <hip/hip_runtime.h>
#include <hip/hip_bf16.h>

#define BATCH     16384
#define SEQ_LEN   40
#define UNITS     256
#define NUM_CHARS 128
#define GATES     1024          // 4 * UNITS
#define MTILE     32            // batch rows per block

// ---- WMMA vector types (probe-confirmed signatures) ----
typedef __attribute__((ext_vector_type(16))) __bf16 v16bf;
typedef __attribute__((ext_vector_type(8)))  __bf16 v8bf;
typedef __attribute__((ext_vector_type(8)))  float  v8f;

// Global-address-space views (keeps loads as global_load after asm laundering)
typedef __bf16 __attribute__((address_space(1))) gbf16;
typedef v16bf  __attribute__((address_space(1))) gv16bf;

union AFrag { v16bf v; struct { v8bf lo; v8bf hi; } p; };

// LDS row stride for the MTILE x 256 h tile (pad 8 halves = 16B to break bank conflicts)
#define HS_STRIDE 264

__device__ __forceinline__ float fast_sigmoid(float x) {
    return __builtin_amdgcn_rcpf(1.0f + __expf(-x));
}
__device__ __forceinline__ float fast_tanh(float x) {
    float e = __expf(-2.0f * x);
    return (1.0f - e) * __builtin_amdgcn_rcpf(1.0f + e);
}

// -------------------- pack W_hh (f32 row-major [256,1024]) into bf16 B-fragments ----
// Block (nTile, kb): 32x16 bf16 block. Element (lane, j):
//   k = kb*32 + (j<8 ? (lane/16)*8 + j : 16 + (lane/16)*8 + j-8),  n = nTile*16 + (lane%16)
// stored at Wp[(nTile*8 + kb)*512 + lane*16 + j]  -> GEMM reads one v16bf per lane.
__global__ void pack_whh(const float* __restrict__ W_hh, __bf16* __restrict__ Wp) {
    int nTile = blockIdx.x >> 3;
    int kb    = blockIdx.x & 7;
    int lane  = threadIdx.x;            // 0..31
    int half  = lane >> 4;
    int m     = lane & 15;
    __bf16* dst = Wp + ((size_t)(nTile * 8 + kb)) * 512 + (size_t)lane * 16;
    int n = nTile * 16 + m;
#pragma unroll
    for (int j = 0; j < 16; ++j) {
        int k = kb * 32 + (j < 8 ? half * 8 + j : 16 + half * 8 + (j - 8));
        dst[j] = (__bf16)W_hh[(size_t)k * GATES + n];
    }
}

// -------------------- fused LSTM: all 40 timesteps + output projection + softmax ----
// grid = BATCH/MTILE blocks, 256 threads (8 waves). Block owns batch rows
// [MTILE*bid, +MTILE). h lives in LDS, c lives in registers, for the whole sequence.
// Wave w owns units [w*32, w*32+32): per step, 2 M-subtiles x {i,f,g,o} x {sub0,sub1}
// = 16 WMMA accumulator tiles; each B fragment feeds 2 WMMAs.
__global__ void __launch_bounds__(256)
lstm_fused(const int*    __restrict__ inputs,
           const float*  __restrict__ W_ih,
           const float*  __restrict__ bias,
           const __bf16* __restrict__ Wp,
           const float*  __restrict__ W_out,
           const float*  __restrict__ b_out,
           float*        __restrict__ out) {
    __shared__ __align__(16) __bf16 hs[MTILE * HS_STRIDE];   // 16.9 KB
    __shared__ int   toks[MTILE * SEQ_LEN];                  // 5 KB
    __shared__ float zsm[MTILE][NUM_CHARS];                  // 16 KB
    __shared__ float red[256];                               // 1 KB

    const int tid  = threadIdx.x;
    const int wave = tid >> 5;
    const int lane = tid & 31;
    const int half = lane >> 4;
    const int m    = lane & 15;
    const int rowBase = blockIdx.x * MTILE;

    // ---- one-time staging: zero h tile, load this block's tokens ----
    for (int i = tid; i < MTILE * HS_STRIDE; i += 256) hs[i] = (__bf16)0.0f;
    for (int i = tid; i < MTILE * SEQ_LEN; i += 256)
        toks[i] = inputs[(size_t)(rowBase + i / SEQ_LEN) * SEQ_LEN + (i % SEQ_LEN)];

    // cell state in registers: cr[st*16 + s*8 + r]
    float cr[32];
#pragma unroll
    for (int i = 0; i < 32; ++i) cr[i] = 0.0f;

    __syncthreads();

    // Global-AS view of the packed weights, laundered once per time step:
    //  - address_space(1) keeps the loads as global_load (LOADcnt only; flat would
    //    also tick DScnt and force full loadcnt+dscnt drains before every WMMA)
    //  - the opaque asm blocks LICM from hoisting 64 B-fragments (512 VGPRs ->
    //    scratch spills, as seen two rounds ago)
#pragma clang diagnostic push
#pragma clang diagnostic ignored "-Wold-style-cast"
    const gbf16* WpG = (const gbf16*)Wp;
#pragma clang diagnostic pop

    // =================== time loop (rolled) ===================
#pragma unroll 1
    for (int t = 0; t < SEQ_LEN; ++t) {
        asm volatile("" : "+s"(WpG));   // opaque: block cross-iteration hoist/CSE

        v8f acc[16] = {};   // [st*8 + gi*2 + s]

        // ---- GEMM: z-tile = h @ W_hh for this block's 32 rows, wave's 128 cols ----
#pragma unroll
        for (int kb = 0; kb < 8; ++kb) {
            // A fragments for both M-subtiles (rows m and m+16 of the h tile)
            AFrag a0, a1;
            {
                const __bf16* r0 = &hs[m * HS_STRIDE + kb * 32 + half * 8];
                const __bf16* r1 = &hs[(m + 16) * HS_STRIDE + kb * 32 + half * 8];
                a0.p.lo = *(const v8bf*)(r0);
                a0.p.hi = *(const v8bf*)(r0 + 16);
                a1.p.lo = *(const v8bf*)(r1);
                a1.p.hi = *(const v8bf*)(r1 + 16);
            }
#pragma unroll
            for (int ti = 0; ti < 8; ++ti) {
                int nTile = (ti >> 1) * 16 + wave * 2 + (ti & 1);
                v16bf bf = *(const gv16bf*)(WpG + ((size_t)(nTile * 8 + kb)) * 512 +
                                            (size_t)lane * 16);
                acc[ti]     = __builtin_amdgcn_wmma_f32_16x16x32_bf16(
                    false, a0.v, false, bf, (short)0, acc[ti],     false, false);
                acc[8 + ti] = __builtin_amdgcn_wmma_f32_16x16x32_bf16(
                    false, a1.v, false, bf, (short)0, acc[8 + ti], false, false);
            }
        }

        __syncthreads();   // all A-fragment reads of hs done before rewriting it

        // ---- fused gate epilogue: all four gates in this wave's registers ----
#pragma unroll
        for (int st = 0; st < 2; ++st) {
#pragma unroll
            for (int s = 0; s < 2; ++s) {
                const int u = wave * 32 + s * 16 + m;      // unit in [0,256)
#pragma unroll
                for (int r = 0; r < 8; ++r) {
                    const int rl = st * 16 + half * 8 + r; // local row 0..31
                    const float* xw = W_ih + (size_t)toks[rl * SEQ_LEN + t] * GATES;

                    float zi = acc[st * 8 + 0 + s][r] + xw[u      ] + bias[u      ];
                    float zf = acc[st * 8 + 2 + s][r] + xw[u + 256] + bias[u + 256];
                    float zg = acc[st * 8 + 4 + s][r] + xw[u + 512] + bias[u + 512];
                    float zo = acc[st * 8 + 6 + s][r] + xw[u + 768] + bias[u + 768];

                    float iv = fast_sigmoid(zi);
                    float fv = fast_sigmoid(zf);
                    float gv = fast_tanh(zg);
                    float ov = fast_sigmoid(zo);

                    float cn = fv * cr[st * 16 + s * 8 + r] + iv * gv;
                    cr[st * 16 + s * 8 + r] = cn;
                    hs[rl * HS_STRIDE + u] = (__bf16)(ov * fast_tanh(cn));
                }
            }
        }

        __syncthreads();   // h tile fully updated before next step's reads
    }

    // =================== fused output: logits = h @ W_out + b_out, softmax ========
    // 256 threads: group g = tid/128 handles rows [g*16, g*16+16), char n = tid%128.
    const int n = tid & 127;
    const int g = tid >> 7;

    float lacc[16];
#pragma unroll
    for (int r = 0; r < 16; ++r) lacc[r] = b_out[n];

    for (int k = 0; k < UNITS; ++k) {
        float wv = W_out[(size_t)k * NUM_CHARS + n];
#pragma unroll
        for (int r = 0; r < 16; ++r)
            lacc[r] += (float)hs[(g * 16 + r) * HS_STRIDE + k] * wv;
    }
#pragma unroll
    for (int r = 0; r < 16; ++r) zsm[g * 16 + r][n] = lacc[r];
    __syncthreads();

    for (int r = 0; r < 16; ++r) {
        const int row = g * 16 + r;
        red[tid] = zsm[row][n];
        __syncthreads();
        for (int s = 64; s > 0; s >>= 1) {
            if (n < s) red[tid] = fmaxf(red[tid], red[tid + s]);
            __syncthreads();
        }
        float mx = red[g * 128];
        __syncthreads();
        float e = __expf(zsm[row][n] - mx);
        red[tid] = e;
        __syncthreads();
        for (int s = 64; s > 0; s >>= 1) {
            if (n < s) red[tid] += red[tid + s];
            __syncthreads();
        }
        float sum = red[g * 128];
        __syncthreads();
        out[(size_t)(rowBase + row) * NUM_CHARS + n] = e / sum;
    }
}

// -------------------- host launcher --------------------
extern "C" void kernel_launch(void* const* d_in, const int* in_sizes, int n_in,
                              void* d_out, int out_size, void* d_ws, size_t ws_size,
                              hipStream_t stream) {
    const int*   inputs = (const int*)  d_in[0];   // [B, T] int32
    const float* W_ih   = (const float*)d_in[1];   // [128, 1024]
    const float* W_hh   = (const float*)d_in[2];   // [256, 1024]
    const float* b      = (const float*)d_in[3];   // [1024]
    const float* W_out  = (const float*)d_in[4];   // [256, 128]
    const float* b_out  = (const float*)d_in[5];   // [128]
    float* out = (float*)d_out;

    // workspace: packed W_hh bf16 (512 KB)
    __bf16* Wp = (__bf16*)d_ws;

    pack_whh<<<64 * 8, 32, 0, stream>>>(W_hh, Wp);
    lstm_fused<<<BATCH / MTILE, 256, 0, stream>>>(inputs, W_ih, b, Wp,
                                                  W_out, b_out, out);
}
